// RNN2_73538430042577
// MI455X (gfx1250) — matrix operand order repaired and checked by reading, not compiled, not run on hardware
//
#include <hip/hip_runtime.h>
#include <hip/hip_bf16.h>
#include <math.h>
#include <stdint.h>

// ---------------------------------------------------------------------------
// Problem constants
// ---------------------------------------------------------------------------
#define HDIM   1024
#define FDIM   256
#define LSEQ   4096
#define NGATE  4096          // 4*H
#define NSTEP  4095          // scan over features[:-1]

// scan persistent-kernel geometry
#define GWG    128           // persistent workgroups (one per WGP, LDS-forced)
#define UPW    8             // hidden units owned per WG per layer
#define ROWS   32            // gate rows per layer per WG (4 gates * UPW)
#define WPAD   513           // dword pitch of one 1024-wide bf16 weight row (bank-conflict pad)

typedef __attribute__((ext_vector_type(16))) _Float16 v16h;
typedef __attribute__((ext_vector_type(8)))  _Float16 v8h;
typedef __attribute__((ext_vector_type(4)))  _Float16 v4h;
typedef __attribute__((ext_vector_type(8)))  float    v8f;

// ---------------------------------------------------------------------------
// small helpers
// ---------------------------------------------------------------------------
__device__ __forceinline__ uint32_t f2bf(float f) {
  uint32_t b = __float_as_uint(f);
  return (b + 0x7fffu + ((b >> 16) & 1u)) >> 16;     // RNE to bf16
}
__device__ __forceinline__ uint32_t pack_bf2(float lo, float hi) {
  return f2bf(lo) | (f2bf(hi) << 16);
}
__device__ __forceinline__ float bf_lo(uint32_t u) { return __uint_as_float(u << 16); }
__device__ __forceinline__ float bf_hi(uint32_t u) { return __uint_as_float(u & 0xffff0000u); }
__device__ __forceinline__ float sigmoidf_(float x) { return 1.0f / (1.0f + expf(-x)); }

// device-scope grid barrier: monotonically increasing counter, never reset
__device__ __forceinline__ void grid_sync(unsigned* cnt, unsigned expected) {
  __syncthreads();
  if (threadIdx.x == 0) {
    __hip_atomic_fetch_add(cnt, 1u, __ATOMIC_RELEASE, __HIP_MEMORY_SCOPE_AGENT);
    while (__hip_atomic_load(cnt, __ATOMIC_ACQUIRE, __HIP_MEMORY_SCOPE_AGENT) < expected)
      __builtin_amdgcn_s_sleep(2);
  }
  __syncthreads();
}

// ---------------------------------------------------------------------------
// conversion kernels (fp32 -> f16 operands for the WMMA GEMM)
// ---------------------------------------------------------------------------
__global__ void conv_feat(const float* __restrict__ f, _Float16* __restrict__ o) {
  int i = blockIdx.x * 256 + threadIdx.x;            // L*F = 1,048,576 elements
  o[i] = (_Float16)f[i];
}

// W_ih1 is (4096 x 257) row-major; build Wt (256 x 4096) with Wt[k][j] = W[j][k]
__global__ void conv_w1t(const float* __restrict__ W, _Float16* __restrict__ Wt) {
  int i = blockIdx.x * 256 + threadIdx.x;            // 256*4096 elements
  int k = i >> 12;
  int j = i & 4095;
  Wt[i] = (_Float16)W[j * 257 + k];
}

__global__ void init_state(float* predacc, unsigned* cnt) {
  if (threadIdx.x == 0) {
    predacc[0] = 0.0f;
    predacc[1] = 0.0f;
    *cnt = 0u;
  }
}

// ---------------------------------------------------------------------------
// Phase 1: XP[t][row] = sum_k features[t][k] * W_ih1[row][k]  (K = 256)
// WMMA f16 GEMM:  C(4096x4096) = A(4096x256) * B(256x4096)
// macro tile 64(M) x 32(N), 8 waves (wm 0..3, wn 0..1), Ktile 32,
// double-buffered LDS: one barrier per K-step, global loads overlap v_wmma.
// ---------------------------------------------------------------------------
#define APAD 40   // f16 pitch pad: 80B rows -> conflict-free 16B lane reads

__global__ __launch_bounds__(256) void xproj_gemm(const _Float16* __restrict__ Ah,
                                                  const _Float16* __restrict__ Bt,
                                                  float* __restrict__ XP) {
  __shared__ _Float16 Ash[2][64][APAD];    // A tiles, row-major (m,k)
  __shared__ _Float16 BshT[2][32][APAD];   // B tiles stored transposed: [n][k]

  const int tid  = threadIdx.x;
  const int m0   = blockIdx.y * 64;
  const int n0   = blockIdx.x * 32;
  const int wave = tid >> 5;
  const int lane = tid & 31;
  const int wm   = wave >> 1;            // 0..3
  const int wn   = wave & 1;             // 0..1
  const int half = lane >> 4;            // lane hi/lo group
  const int l16  = lane & 15;

  v8f acc = {0.f, 0.f, 0.f, 0.f, 0.f, 0.f, 0.f, 0.f};

  const int ar = (tid * 8) >> 5, ac = (tid * 8) & 31;   // A staging: 8 f16/thread
  const int br = (tid * 4) >> 5, bc = (tid * 4) & 31;   // B staging: 4 f16/thread

  // prologue: stage k0 = 0 into buffer 0
  {
    *(v8h*)&Ash[0][ar][ac] = *(const v8h*)&Ah[(size_t)(m0 + ar) * FDIM + ac];
    v4h bv = *(const v4h*)&Bt[(size_t)br * NGATE + n0 + bc];
    BshT[0][bc + 0][br] = bv[0];
    BshT[0][bc + 1][br] = bv[1];
    BshT[0][bc + 2][br] = bv[2];
    BshT[0][bc + 3][br] = bv[3];
  }

  const int mr = wm * 16 + l16;
  const int nc = wn * 16 + l16;

  for (int kt = 0; kt < FDIM / 32; ++kt) {
    __syncthreads();                       // staging of buf `cur` complete,
    const int cur = kt & 1;                // readers of buf `cur^1` done

    // A fragment (16x32 f16):
    //   lane<16 : a[0..7]=K0..7,  a[8..15]=K16..23
    //   lane>=16: a[0..7]=K8..15, a[8..15]=K24..31
    v8h alo = *(const v8h*)&Ash[cur][mr][half * 8];
    v8h ahi = *(const v8h*)&Ash[cur][mr][16 + half * 8];
    v16h a = __builtin_shufflevector(alo, ahi, 0,1,2,3,4,5,6,7,8,9,10,11,12,13,14,15);

    // B fragment (32x16 f16): lane<16 -> K0..7,K16..23 of column l16; lane>=16 -> K8..15,K24..31
    v8h blo = *(const v8h*)&BshT[cur][nc][half * 8];
    v8h bhi = *(const v8h*)&BshT[cur][nc][16 + half * 8];
    v16h b = __builtin_shufflevector(blo, bhi, 0,1,2,3,4,5,6,7,8,9,10,11,12,13,14,15);

    // stage next tile into the other buffer while WMMA consumes registers
    if (kt < FDIM / 32 - 1) {
      const int k0n = (kt + 1) * 32;
      const int nxt = cur ^ 1;
      *(v8h*)&Ash[nxt][ar][ac] = *(const v8h*)&Ah[(size_t)(m0 + ar) * FDIM + k0n + ac];
      v4h bv = *(const v4h*)&Bt[(size_t)(k0n + br) * NGATE + n0 + bc];
      BshT[nxt][bc + 0][br] = bv[0];
      BshT[nxt][bc + 1][br] = bv[1];
      BshT[nxt][bc + 2][br] = bv[2];
      BshT[nxt][bc + 3][br] = bv[3];
    }

    acc = __builtin_amdgcn_wmma_f32_16x16x32_f16(false, a, false, b,
                                                 (short)0, acc, false, false);
  }

  // C layout: VGPR r -> M = wmbase + half*8 + r, N = l16
  const int mb = m0 + wm * 16 + half * 8;
  const int nb = n0 + wn * 16 + l16;
#pragma unroll
  for (int r = 0; r < 8; ++r)
    XP[(size_t)(mb + r) * NGATE + nb] = acc[r];
}

// ---------------------------------------------------------------------------
// Phase 2: persistent sequential scan.
// 128 WGs x 256 threads. WG g owns hidden units [g*8, g*8+8) of both layers,
// keeps its 96 bf16 weight rows (~197KB) resident in LDS for all 4095 steps.
// ---------------------------------------------------------------------------
__global__ __launch_bounds__(256, 1) void scan_kernel(
    const float* __restrict__ XP,
    const float* __restrict__ W_ih1,   // 4096 x 257 (only col 256 used here)
    const float* __restrict__ b_ih1,
    const float* __restrict__ W_hh1,   // 4096 x 1024
    const float* __restrict__ b_hh1,
    const float* __restrict__ W_ih2,   // 4096 x 1024
    const float* __restrict__ b_ih2,
    const float* __restrict__ W_hh2,   // 4096 x 1024
    const float* __restrict__ b_hh2,
    const float* __restrict__ W_out,   // 1024
    const float* __restrict__ b_out,   // 1
    const float* __restrict__ targets, // 4096
    float* h1g,      // 2 x 1024 (double buffered by step parity)
    float* h2g,      // 2 x 1024
    float* predacc,  // 2
    unsigned* cnt,
    float* out) {
  extern __shared__ char smem[];
  uint32_t* w1  = (uint32_t*)smem;                 // ROWS*WPAD dwords (bf16 pairs)
  uint32_t* w2i = w1  + ROWS * WPAD;
  uint32_t* w2h = w2i + ROWS * WPAD;
  float* ha    = (float*)(w2h + ROWS * WPAD);      // 1024
  float* hb    = ha + HDIM;                        // 1024
  float* red   = hb + HDIM;                        // 256
  float* g32   = red + 256;                        // 32
  float* b1c   = g32 + 32;                         // 32
  float* wpc   = b1c + 32;                         // 32
  float* b2c   = wpc + 32;                         // 32
  float* woutc = b2c + 32;                         // 8
  float* c1s   = woutc + UPW;                      // 8
  float* c2s   = c1s + UPW;                        // 8
  float* h2s   = c2s + UPW;                        // 8

  const int tid = threadIdx.x;
  const int wg  = blockIdx.x;
  const int u0  = wg * UPW;

  // ---- one-time init: pull owned weight rows into LDS as packed bf16 ----
  for (int idx = tid; idx < ROWS * 512; idx += 256) {
    int q  = idx >> 9;          // local gate-row 0..31  (q = gate*8 + unit)
    int kk = idx & 511;         // dword index along K
    int row = (q >> 3) * HDIM + u0 + (q & 7);
    const float* s1 = &W_hh1[(size_t)row * HDIM + 2 * kk];
    const float* s2 = &W_ih2[(size_t)row * HDIM + 2 * kk];
    const float* s3 = &W_hh2[(size_t)row * HDIM + 2 * kk];
    w1 [q * WPAD + kk] = pack_bf2(s1[0], s1[1]);
    w2i[q * WPAD + kk] = pack_bf2(s2[0], s2[1]);
    w2h[q * WPAD + kk] = pack_bf2(s3[0], s3[1]);
  }
  if (tid < ROWS) {
    int row = (tid >> 3) * HDIM + u0 + (tid & 7);
    b1c[tid] = b_ih1[row] + b_hh1[row];
    b2c[tid] = b_ih2[row] + b_hh2[row];
    wpc[tid] = W_ih1[(size_t)row * 257 + 256];     // p-column of layer-1 input weights
  }
  if (tid < UPW) {
    woutc[tid] = W_out[u0 + tid];
    c1s[tid] = 0.0f;
    c2s[tid] = 0.0f;
    __hip_atomic_store(&h1g[u0 + tid], 0.0f, __ATOMIC_RELAXED, __HIP_MEMORY_SCOPE_AGENT);
    __hip_atomic_store(&h2g[u0 + tid], 0.0f, __ATOMIC_RELAXED, __HIP_MEMORY_SCOPE_AGENT);
  }
  float p = targets[0];
  const float bo = b_out[0];
  float lossacc = 0.0f;
  unsigned epoch = 0;

  epoch++; grid_sync(cnt, epoch * GWG);            // init visible everywhere

  const int rr = tid & 31;                         // gate-row this thread reduces
  const int cc = tid >> 5;                         // K chunk 0..7 (128 wide)
  const int kb = cc * 128;

  for (int s = 0; s < NSTEP; ++s) {
    const float* h1prev = h1g + (s & 1) * HDIM;
    float*       h1next = h1g + ((s + 1) & 1) * HDIM;
    const float* h2prev = h2g + (s & 1) * HDIM;
    float*       h2next = h2g + ((s + 1) & 1) * HDIM;

    // hide next step's scattered XP reads behind this step's compute
    if (tid < ROWS && s + 1 < NSTEP) {
      int row = (tid >> 3) * HDIM + u0 + (tid & 7);
      __builtin_prefetch(&XP[(size_t)(s + 1) * NGATE + row], 0, 1);
    }

    // ---------------- stage 1: layer-1 gates + activation ----------------
    for (int k = tid; k < HDIM; k += 256)
      ha[k] = __hip_atomic_load(&h1prev[k], __ATOMIC_RELAXED, __HIP_MEMORY_SCOPE_AGENT);
    __syncthreads();
    {
      const uint32_t* wr = &w1[rr * WPAD + (kb >> 1)];
      const float* hv = &ha[kb];
      float acc = 0.0f;
#pragma unroll 8
      for (int kk = 0; kk < 64; ++kk) {
        uint32_t u = wr[kk];
        acc = fmaf(bf_lo(u), hv[2 * kk],     acc);
        acc = fmaf(bf_hi(u), hv[2 * kk + 1], acc);
      }
      red[tid] = acc;
    }
    __syncthreads();
    if (tid < ROWS) {
      float t = 0.0f;
#pragma unroll
      for (int c = 0; c < 8; ++c) t += red[c * 32 + tid];
      int row = (tid >> 3) * HDIM + u0 + (tid & 7);
      g32[tid] = t + XP[(size_t)s * NGATE + row] + b1c[tid] + wpc[tid] * p;
    }
    __syncthreads();
    if (tid < UPW) {
      float gi = sigmoidf_(g32[tid]);
      float gf = sigmoidf_(g32[8 + tid]);
      float gg = tanhf(g32[16 + tid]);
      float go = sigmoidf_(g32[24 + tid]);
      float c  = gf * c1s[tid] + gi * gg;
      c1s[tid] = c;
      float h  = go * tanhf(c);
      __hip_atomic_store(&h1next[u0 + tid], h, __ATOMIC_RELAXED, __HIP_MEMORY_SCOPE_AGENT);
    }
    epoch++; grid_sync(cnt, epoch * GWG);          // h1_t complete everywhere

    // ---------------- stage 2: layer-2 gates + activation + pred ---------
    for (int k = tid; k < HDIM; k += 256) {
      ha[k] = __hip_atomic_load(&h1next[k], __ATOMIC_RELAXED, __HIP_MEMORY_SCOPE_AGENT);
      hb[k] = __hip_atomic_load(&h2prev[k], __ATOMIC_RELAXED, __HIP_MEMORY_SCOPE_AGENT);
    }
    __syncthreads();
    {
      const uint32_t* wi = &w2i[rr * WPAD + (kb >> 1)];
      const uint32_t* wh = &w2h[rr * WPAD + (kb >> 1)];
      const float* hv = &ha[kb];
      const float* gv = &hb[kb];
      float acc = 0.0f;
#pragma unroll 4
      for (int kk = 0; kk < 64; ++kk) {
        uint32_t ui = wi[kk];
        uint32_t uh = wh[kk];
        acc = fmaf(bf_lo(ui), hv[2 * kk],     acc);
        acc = fmaf(bf_hi(ui), hv[2 * kk + 1], acc);
        acc = fmaf(bf_lo(uh), gv[2 * kk],     acc);
        acc = fmaf(bf_hi(uh), gv[2 * kk + 1], acc);
      }
      red[tid] = acc;
    }
    __syncthreads();
    if (tid < ROWS) {
      float t = 0.0f;
#pragma unroll
      for (int c = 0; c < 8; ++c) t += red[c * 32 + tid];
      g32[tid] = t + b2c[tid];
    }
    __syncthreads();
    if (tid < UPW) {
      float gi = sigmoidf_(g32[tid]);
      float gf = sigmoidf_(g32[8 + tid]);
      float gg = tanhf(g32[16 + tid]);
      float go = sigmoidf_(g32[24 + tid]);
      float c  = gf * c2s[tid] + gi * gg;
      c2s[tid] = c;
      float h  = go * tanhf(c);
      h2s[tid] = h;
      __hip_atomic_store(&h2next[u0 + tid], h, __ATOMIC_RELAXED, __HIP_MEMORY_SCOPE_AGENT);
    }
    __syncthreads();
    if (tid == 0) {
      float part = 0.0f;
#pragma unroll
      for (int j = 0; j < UPW; ++j) part += woutc[j] * h2s[j];
      __hip_atomic_fetch_add(&predacc[s & 1], part, __ATOMIC_RELAXED, __HIP_MEMORY_SCOPE_AGENT);
    }
    epoch++; grid_sync(cnt, epoch * GWG);          // prediction reduced

    float pred = __hip_atomic_load(&predacc[s & 1], __ATOMIC_RELAXED,
                                   __HIP_MEMORY_SCOPE_AGENT) + bo + p;
    if (wg == 0 && tid == 0) {
      // re-arm the other slot for step s+1 (safe: before our sync1(s+1) arrival)
      __hip_atomic_store(&predacc[(s + 1) & 1], 0.0f, __ATOMIC_RELAXED,
                         __HIP_MEMORY_SCOPE_AGENT);
      float d = pred - targets[s + 1];
      lossacc += d * d;
    }
    p = pred;
  }

  if (wg == 0 && tid == 0)
    out[0] = sqrtf(lossacc * (1.0f / (float)LSEQ));
}

// ---------------------------------------------------------------------------
// host launch
// ---------------------------------------------------------------------------
extern "C" void kernel_launch(void* const* d_in, const int* in_sizes, int n_in,
                              void* d_out, int out_size, void* d_ws, size_t ws_size,
                              hipStream_t stream) {
  (void)in_sizes; (void)n_in; (void)out_size; (void)ws_size;
  const float* features = (const float*)d_in[0];
  const float* targets  = (const float*)d_in[1];
  const float* W_ih1    = (const float*)d_in[2];
  const float* b_ih1    = (const float*)d_in[3];
  const float* W_hh1    = (const float*)d_in[4];
  const float* b_hh1    = (const float*)d_in[5];
  const float* W_ih2    = (const float*)d_in[6];
  const float* b_ih2    = (const float*)d_in[7];
  const float* W_hh2    = (const float*)d_in[8];
  const float* b_hh2    = (const float*)d_in[9];
  const float* W_out    = (const float*)d_in[10];
  const float* b_out    = (const float*)d_in[11];
  float* out = (float*)d_out;

  char* ws = (char*)d_ws;
  float*    XP      = (float*)   (ws + 0);                 // 4096*4096*4 = 64 MB
  _Float16* Fh      = (_Float16*)(ws + 67108864);          // 2 MB
  _Float16* Wt      = (_Float16*)(ws + 69206016);          // 2 MB
  float*    h1g     = (float*)   (ws + 71303168);          // 8 KB
  float*    h2g     = (float*)   (ws + 71311360);          // 8 KB
  float*    predacc = (float*)   (ws + 71319552);          // 8 B
  unsigned* cnt     = (unsigned*)(ws + 71319616);          // 4 B

  // operand conversion + state init
  conv_feat<<<(LSEQ * FDIM) / 256, 256, 0, stream>>>(features, Fh);
  conv_w1t<<<(FDIM * NGATE) / 256, 256, 0, stream>>>(W_ih1, Wt);
  init_state<<<1, 64, 0, stream>>>(predacc, cnt);

  // phase 1: WMMA GEMM for all input projections
  dim3 gg(NGATE / 32, LSEQ / 64);
  xproj_gemm<<<gg, 256, 0, stream>>>(Fh, Wt, XP);

  // phase 2: persistent LSTM scan, weights resident in LDS
  size_t lds = (size_t)3 * ROWS * WPAD * 4                  // packed bf16 weights
             + (size_t)(HDIM * 2 + 256 + 32 * 3 + UPW * 4) * 4;
  hipFuncSetAttribute((const void*)scan_kernel,
                      hipFuncAttributeMaxDynamicSharedMemorySize, (int)lds);
  scan_kernel<<<GWG, 256, lds, stream>>>(XP, W_ih1, b_ih1, W_hh1, b_hh1,
                                         W_ih2, b_ih2, W_hh2, b_hh2,
                                         W_out, b_out, targets,
                                         h1g, h2g, predacc, cnt, out);
}
